// DecoderLayer_21191368638583
// MI455X (gfx1250) — compile-verified, matrix operand order checked by reading, not run on hardware
//
#include <hip/hip_runtime.h>
#include <hip/hip_bf16.h>

// ---------------------------------------------------------------------------
// Transformer decoder layer for MI455X (gfx1250), wave32 + WMMA.
// All heavy GEMMs run on v_wmma_f32_16x16x32_f16 (f16 inputs, f32 accum).
// Softmax / layernorm / cross-attn reductions stay in f32.
// ---------------------------------------------------------------------------

typedef __attribute__((ext_vector_type(16))) _Float16 v16h;
typedef __attribute__((ext_vector_type(8)))  _Float16 v8h;
typedef __attribute__((ext_vector_type(8)))  float    v8f;

#define DEV __device__ __forceinline__

// Problem constants
#define CL   512      // target length
#define CS   2048     // memory length
#define CB   4        // batch
#define CE   1024     // embed
#define CH   16       // heads
#define CHD  64       // head dim
#define CFF  2048     // ffn dim

// ---------------------------------------------------------------------------
// WMMA fragment loaders.
// 16-bit A (16x32) ISA layout: lane&15 selects the M (or N) row;
// lanes 0-15 hold K = {k0..k0+7, k0+16..k0+23},
// lanes 16-31 hold K = {k0+8..k0+15, k0+24..k0+31}.  B (32x16) mirrors A with
// the lane index selecting the N column.
// ---------------------------------------------------------------------------
DEV v16h frag_load_contig(const _Float16* __restrict__ base, long row_stride,
                          int lane, int k0) {
  int r  = lane & 15;
  int kh = (lane >> 4) << 3;          // 0 or 8
  const _Float16* p = base + (long)r * row_stride + k0 + kh;
  v8h lo = *(const v8h*)(p);          // 16B vector load
  v8h hi = *(const v8h*)(p + 16);
  v16h out;
#pragma unroll
  for (int i = 0; i < 8; ++i) { out[i] = lo[i]; out[i + 8] = hi[i]; }
  return out;
}

// B stored [K,N] with unit N-stride and arbitrary K-stride (attn @ V case).
DEV v16h frag_load_strided(const _Float16* __restrict__ base, long k_stride,
                           int lane, int k0) {
  int n  = lane & 15;
  int kh = (lane >> 4) << 3;
  const _Float16* p = base + n;
  v16h out;
#pragma unroll
  for (int i = 0; i < 8; ++i) {
    out[i]     = p[(long)(k0 + kh + i) * k_stride];
    out[i + 8] = p[(long)(k0 + 16 + kh + i) * k_stride];
  }
  return out;
}

// ---------------------------------------------------------------------------
// Generic batched GEMM:  C[m,n] = alpha * sum_k A[m,k]*B[k,n] + bias[n]
//   A:  element at A + m*a_rs + k            (k contiguous)
//   B:  BKC=true : element at Bm + n*b_s + k (weight W[N,K] row-major, k contig)
//       BKC=false: element at Bm + k*b_s + n (value  V[K,N],  n contig)
//   batch z: zo=z/zdiv, zi=z%zdiv; each operand offset by zo*X_zo + zi*X_zi.
// Block: 256 threads = 8 waves as 4x2, wave tile 32x32, block tile 128x64.
// Requires M%128==0, N%64==0, K%32==0 (true for every call here).
// ---------------------------------------------------------------------------
template <bool BKC>
__global__ __launch_bounds__(256) void gemm_wmma(
    const _Float16* __restrict__ A, long a_rs, long a_zo, long a_zi,
    const _Float16* __restrict__ Bm, long b_s, long b_zo, long b_zi,
    const float* __restrict__ bias, float alpha, int relu,
    float* __restrict__ C32, _Float16* __restrict__ C16,
    long c_rs, long c_zo, long c_zi,
    int M, int N, int K, int zdiv) {
  int z  = blockIdx.z;
  int zo = z / zdiv, zi = z % zdiv;
  A  += (long)zo * a_zo + (long)zi * a_zi;
  Bm += (long)zo * b_zo + (long)zi * b_zi;
  long coff = (long)zo * c_zo + (long)zi * c_zi;

  int lane = threadIdx.x & 31;
  int wave = threadIdx.x >> 5;
  int wm   = wave >> 1;                       // 0..3
  int wn   = wave & 1;                        // 0..1
  int m0   = blockIdx.x * 128 + wm * 32;
  int n0   = blockIdx.y * 64  + wn * 32;

  const v8f zero = {0.f, 0.f, 0.f, 0.f, 0.f, 0.f, 0.f, 0.f};
  v8f acc[2][2];
#pragma unroll
  for (int i = 0; i < 2; ++i)
#pragma unroll
    for (int j = 0; j < 2; ++j) acc[i][j] = zero;

  const _Float16* Ab0 = A + (long)m0 * a_rs;
  const _Float16* Ab1 = A + (long)(m0 + 16) * a_rs;

  for (int k0 = 0; k0 < K; k0 += 32) {
    v16h a0 = frag_load_contig(Ab0, a_rs, lane, k0);
    v16h a1 = frag_load_contig(Ab1, a_rs, lane, k0);
    v16h b0, b1;
    if (BKC) {
      b0 = frag_load_contig(Bm + (long)n0 * b_s,        b_s, lane, k0);
      b1 = frag_load_contig(Bm + (long)(n0 + 16) * b_s, b_s, lane, k0);
    } else {
      b0 = frag_load_strided(Bm + n0,      b_s, lane, k0);
      b1 = frag_load_strided(Bm + n0 + 16, b_s, lane, k0);
    }
    acc[0][0] = __builtin_amdgcn_wmma_f32_16x16x32_f16(false, a0, false, b0,
                                                       (short)0, acc[0][0], false, false);
    acc[0][1] = __builtin_amdgcn_wmma_f32_16x16x32_f16(false, a0, false, b1,
                                                       (short)0, acc[0][1], false, false);
    acc[1][0] = __builtin_amdgcn_wmma_f32_16x16x32_f16(false, a1, false, b0,
                                                       (short)0, acc[1][0], false, false);
    acc[1][1] = __builtin_amdgcn_wmma_f32_16x16x32_f16(false, a1, false, b1,
                                                       (short)0, acc[1][1], false, false);
  }

  // C/D layout: N = lane&15, VGPR r holds M = r (+8 for lanes 16-31).
  int nlane = lane & 15;
  int mrow  = (lane >> 4) * 8;
#pragma unroll
  for (int tm = 0; tm < 2; ++tm)
#pragma unroll
    for (int tn = 0; tn < 2; ++tn) {
      int ncol = n0 + tn * 16 + nlane;
      float bv = bias ? bias[ncol] : 0.f;
#pragma unroll
      for (int r = 0; r < 8; ++r) {
        int   mm  = m0 + tm * 16 + mrow + r;
        float v   = acc[tm][tn][r] * alpha + bv;
        if (relu) v = v > 0.f ? v : 0.f;
        long  idx = coff + (long)mm * c_rs + ncol;
        if (C32) C32[idx] = v;
        if (C16) C16[idx] = (_Float16)v;
      }
    }
}

// ---------------------------------------------------------------------------
// Elementwise helpers
// ---------------------------------------------------------------------------
__global__ __launch_bounds__(256) void k_cvt(const float* __restrict__ x,
                                             _Float16* __restrict__ y, long n) {
  long i = (long)blockIdx.x * 256 + threadIdx.x;
  if (i < n) y[i] = (_Float16)x[i];
}

__global__ __launch_bounds__(256) void k_add2(const float* __restrict__ a,
                                              const float* __restrict__ b,
                                              _Float16* __restrict__ y, long n) {
  long i = (long)blockIdx.x * 256 + threadIdx.x;
  if (i < n) y[i] = (_Float16)(a[i] + b[i]);
}

// ---------------------------------------------------------------------------
// Self-attention row softmax: 512-wide rows, f32 in, f16 probs out.
// ---------------------------------------------------------------------------
__global__ __launch_bounds__(256) void k_softmax512(const float* __restrict__ x,
                                                    _Float16* __restrict__ y) {
  long base = (long)blockIdx.x * CL;
  int  tid  = threadIdx.x;
  float v0 = x[base + tid];
  float v1 = x[base + tid + 256];
  __shared__ float red[256];
  red[tid] = fmaxf(v0, v1);
  __syncthreads();
  for (int o = 128; o > 0; o >>= 1) {
    if (tid < o) red[tid] = fmaxf(red[tid], red[tid + o]);
    __syncthreads();
  }
  float mx = red[0];
  __syncthreads();
  float e0 = __expf(v0 - mx), e1 = __expf(v1 - mx);
  red[tid] = e0 + e1;
  __syncthreads();
  for (int o = 128; o > 0; o >>= 1) {
    if (tid < o) red[tid] += red[tid + o];
    __syncthreads();
  }
  float inv = 1.f / red[0];
  y[base + tid]       = (_Float16)(e0 * inv);
  y[base + tid + 256] = (_Float16)(e1 * inv);
}

// ---------------------------------------------------------------------------
// Cross-attention scalar stages (all f32).
// Channel mapping: E index = d*H + h  (reshape (B,E,N)->(B,HD,H,N)).
// Token t of a [N,B,E] tensor lives at row n*B+b.
// ---------------------------------------------------------------------------
// qcsum[b*E+e] = sum_l qc[(l*B+b)*E + e]
__global__ __launch_bounds__(256) void k_qcsum(const float* __restrict__ qc,
                                               float* __restrict__ out) {
  int idx = blockIdx.x * 256 + threadIdx.x;     // < B*E = 4096
  int b = idx >> 10, e = idx & 1023;
  float s = 0.f;
  for (int l = 0; l < CL; ++l) s += qc[(long)l * (CB * CE) + b * CE + e];
  out[idx] = s;
}

// s1[(b*H+h)*S+m] = scale * sum_d qcsum[b*E + d*H+h] * kc[(m*B+b)*E + d*H+h]
__global__ __launch_bounds__(256) void k_s1(const float* __restrict__ qcsum,
                                            const float* __restrict__ kc,
                                            float* __restrict__ s1, float scale) {
  int idx = blockIdx.x * 256 + threadIdx.x;     // < B*H*S = 131072
  int m = idx & (CS - 1);
  int h = (idx >> 11) & (CH - 1);
  int b = idx >> 15;
  const float* kcp = kc + (long)m * (CB * CE) + b * CE + h;
  const float* qsp = qcsum + b * CE + h;
  float s = 0.f;
#pragma unroll 4
  for (int d = 0; d < CHD; ++d) s += qsp[d * CH] * kcp[d * CH];
  s1[idx] = s * scale;
}

// prob[((b*HD+d)*H+h)*S + m] = softmax_m( kc[(m*B+b)*E + d*H+h] * s1[(b*H+h)*S+m] )
__global__ __launch_bounds__(256) void k_cross_softmax(const float* __restrict__ kc,
                                                       const float* __restrict__ s1,
                                                       float* __restrict__ prob) {
  int row = blockIdx.x;                         // (b*HD+d)*H + h, < 4096
  int h = row & (CH - 1);
  int d = (row >> 4) & (CHD - 1);
  int b = row >> 10;
  const float* s1r  = s1 + (long)(b * CH + h) * CS;
  long         kcb  = (long)b * CE + d * CH + h;
  int          tid  = threadIdx.x;
  float loc[8];
  float mx = -3.402823466e38f;
#pragma unroll
  for (int i = 0; i < 8; ++i) {
    int c = tid + i * 256;
    float v = kc[(long)c * (CB * CE) + kcb] * s1r[c];
    loc[i] = v;
    mx = fmaxf(mx, v);
  }
  __shared__ float red[256];
  red[tid] = mx;
  __syncthreads();
  for (int o = 128; o > 0; o >>= 1) {
    if (tid < o) red[tid] = fmaxf(red[tid], red[tid + o]);
    __syncthreads();
  }
  mx = red[0];
  __syncthreads();
  float s = 0.f;
#pragma unroll
  for (int i = 0; i < 8; ++i) {
    loc[i] = __expf(loc[i] - mx);
    s += loc[i];
  }
  red[tid] = s;
  __syncthreads();
  for (int o = 128; o > 0; o >>= 1) {
    if (tid < o) red[tid] += red[tid + o];
    __syncthreads();
  }
  float inv = 1.f / red[0];
  float* pr = prob + (long)row * CS;
#pragma unroll
  for (int i = 0; i < 8; ++i) pr[tid + i * 256] = loc[i] * inv;
}

// w2[(b*H+h)*S+m] = sum_n down_w[n] * prob[((b*HD+n)*H+h)*S + m]
__global__ __launch_bounds__(256) void k_w2(const float* __restrict__ dw,
                                            const float* __restrict__ prob,
                                            float* __restrict__ w2) {
  int idx = blockIdx.x * 256 + threadIdx.x;     // < B*H*S
  int m = idx & (CS - 1);
  int h = (idx >> 11) & (CH - 1);
  int b = idx >> 15;
  const float* pp = prob + (((long)b * CHD * CH + h) * CS) + m;
  float s = 0.f;
#pragma unroll 4
  for (int n = 0; n < CHD; ++n) s += dw[n] * pp[(long)n * CH * CS];
  w2[idx] = s;
}

// od[b*E + d*H+h] = down_b + sum_m w2[(b*H+h)*S+m] * vc[(m*B+b)*E + d*H+h]
__global__ __launch_bounds__(256) void k_od(const float* __restrict__ w2,
                                            const float* __restrict__ vc,
                                            const float* __restrict__ db,
                                            float* __restrict__ od) {
  int row = blockIdx.x;                         // b*E + c, < 4096
  int b = row >> 10;
  int c = row & 1023;                           // c = d*H + h
  int h = c & (CH - 1);
  const float* w2r = w2 + (long)(b * CH + h) * CS;
  long vcb = (long)b * CE + c;
  int tid = threadIdx.x;
  float s = 0.f;
  for (int m = tid; m < CS; m += 256)
    s += w2r[m] * vc[(long)m * (CB * CE) + vcb];
  __shared__ float red[256];
  red[tid] = s;
  __syncthreads();
  for (int o = 128; o > 0; o >>= 1) {
    if (tid < o) red[tid] += red[tid + o];
    __syncthreads();
  }
  if (tid == 0) od[row] = red[0] + db[0];
}

// ---------------------------------------------------------------------------
// LayerNorm over E=1024 with optional residual and optional [B,E] broadcast.
// x = a + (r ? r : 0) + (bc ? bc[token%B] : 0); out32 = LN(x); out16 optional.
// ---------------------------------------------------------------------------
__global__ __launch_bounds__(256) void k_ln(const float* __restrict__ a,
                                            const float* __restrict__ r,
                                            const float* __restrict__ bc,
                                            const float* __restrict__ g,
                                            const float* __restrict__ be,
                                            float* __restrict__ o32,
                                            _Float16* __restrict__ o16) {
  int  t    = blockIdx.x;
  long base = (long)t * CE;
  const float* bcr = bc ? bc + (long)(t % CB) * CE : nullptr;
  int tid = threadIdx.x;
  float x[4];
  float s = 0.f;
#pragma unroll
  for (int i = 0; i < 4; ++i) {
    int   e = tid + i * 256;
    float v = a[base + e];
    if (r)   v += r[base + e];
    if (bcr) v += bcr[e];
    x[i] = v;
    s += v;
  }
  __shared__ float red[256];
  red[tid] = s;
  __syncthreads();
  for (int o = 128; o > 0; o >>= 1) {
    if (tid < o) red[tid] += red[tid + o];
    __syncthreads();
  }
  float mean = red[0] * (1.f / CE);
  __syncthreads();
  float vs = 0.f;
#pragma unroll
  for (int i = 0; i < 4; ++i) {
    float d = x[i] - mean;
    vs += d * d;
  }
  red[tid] = vs;
  __syncthreads();
  for (int o = 128; o > 0; o >>= 1) {
    if (tid < o) red[tid] += red[tid + o];
    __syncthreads();
  }
  float inv = rsqrtf(red[0] * (1.f / CE) + 1e-5f);
#pragma unroll
  for (int i = 0; i < 4; ++i) {
    int   e = tid + i * 256;
    float y = (x[i] - mean) * inv * g[e] + be[e];
    o32[base + e] = y;
    if (o16) o16[base + e] = (_Float16)y;
  }
}

// ---------------------------------------------------------------------------
// Host side
// ---------------------------------------------------------------------------
extern "C" void kernel_launch(void* const* d_in, const int* in_sizes, int n_in,
                              void* d_out, int out_size, void* d_ws, size_t ws_size,
                              hipStream_t stream) {
  const float* tgt  = (const float*)d_in[0];
  const float* mem  = (const float*)d_in[1];
  const float* pos  = (const float*)d_in[2];
  const float* qpos = (const float*)d_in[3];
  const float* inW  = (const float*)d_in[4];
  const float* inB  = (const float*)d_in[5];
  const float* outW = (const float*)d_in[6];
  const float* outB = (const float*)d_in[7];
  const float* cqW  = (const float*)d_in[8];
  const float* cqB  = (const float*)d_in[9];
  const float* ckW  = (const float*)d_in[10];
  const float* ckB  = (const float*)d_in[11];
  const float* cvW  = (const float*)d_in[12];
  const float* cvB  = (const float*)d_in[13];
  const float* dwW  = (const float*)d_in[14];
  const float* dwB  = (const float*)d_in[15];
  const float* l1W  = (const float*)d_in[16];
  const float* l1B  = (const float*)d_in[17];
  const float* l2W  = (const float*)d_in[18];
  const float* l2B  = (const float*)d_in[19];
  const float* n1g  = (const float*)d_in[20];
  const float* n1b  = (const float*)d_in[21];
  const float* n2g  = (const float*)d_in[22];
  const float* n2b  = (const float*)d_in[23];
  const float* n3g  = (const float*)d_in[24];
  const float* n3b  = (const float*)d_in[25];

  const long TOK  = (long)CL * CB;   // 2048 target tokens
  const long MTOK = (long)CS * CB;   // 8192 memory tokens
  const float scale = 0.125f;        // 1/sqrt(64)

  char* ws = (char*)d_ws;
  const size_t MB = 1u << 20;

  // Persistent f16 weights (converted every launch; deterministic).
  _Float16* w_inproj = (_Float16*)(ws + 0 * MB);    // [3E,E]  6MB
  _Float16* w_out    = (_Float16*)(ws + 6 * MB);    // [E,E]   2MB
  _Float16* w_cq     = (_Float16*)(ws + 8 * MB);
  _Float16* w_ck     = (_Float16*)(ws + 10 * MB);
  _Float16* w_cv     = (_Float16*)(ws + 12 * MB);
  _Float16* w_l1     = (_Float16*)(ws + 14 * MB);   // [FF,E]  4MB
  _Float16* w_l2     = (_Float16*)(ws + 18 * MB);   // [E,FF]  4MB
  float*    tgt1     = (float*)(ws + 22 * MB);      // 8MB
  float*    tgt2     = (float*)(ws + 30 * MB);      // 8MB
  char*     SA       = ws + 38 * MB;                // phase-aliased arena

  // Phase A (self-attention)
  _Float16* qkin16  = (_Float16*)(SA + 0 * MB);
  _Float16* tgt16   = (_Float16*)(SA + 4 * MB);
  _Float16* q16     = (_Float16*)(SA + 8 * MB);
  _Float16* k16     = (_Float16*)(SA + 12 * MB);
  _Float16* v16     = (_Float16*)(SA + 16 * MB);
  float*    scores  = (float*)(SA + 20 * MB);       // [B,H,L,L] 64MB
  _Float16* attn16  = (_Float16*)(SA + 84 * MB);    // 32MB
  _Float16* ctx16   = (_Float16*)(SA + 116 * MB);   // 4MB
  float*    selfout = (float*)(SA + 120 * MB);      // 8MB
  // Phase B (cross-attention) — aliases phase A
  _Float16* qin2    = (_Float16*)(SA + 0 * MB);
  float*    qc32    = (float*)(SA + 4 * MB);        // 8MB
  _Float16* mp16    = (_Float16*)(SA + 12 * MB);    // 16MB
  _Float16* mem16   = (_Float16*)(SA + 28 * MB);    // 16MB
  float*    kc32    = (float*)(SA + 44 * MB);       // 32MB
  float*    vc32    = (float*)(SA + 76 * MB);       // 32MB
  float*    prob    = (float*)(SA + 108 * MB);      // 32MB
  float*    qcsum   = (float*)(SA + 140 * MB);
  float*    s1sum   = (float*)(SA + 141 * MB);
  float*    w2buf   = (float*)(SA + 142 * MB);
  float*    odbuf   = (float*)(SA + 143 * MB);
  // Phase C (FFN) — aliases dead phase B regions
  _Float16* tgt2_16 = (_Float16*)(SA + 0 * MB);
  _Float16* h1_16   = (_Float16*)(SA + 4 * MB);     // [TOK,FF] 8MB
  float*    ff32    = (float*)(SA + 12 * MB);       // 8MB

  auto cvt = [&](const float* x, _Float16* y, long n) {
    k_cvt<<<dim3((unsigned)((n + 255) / 256)), 256, 0, stream>>>(x, y, n);
  };

  // ---- weight conversion ----
  cvt(inW,  w_inproj, 3L * CE * CE);
  cvt(outW, w_out,    (long)CE * CE);
  cvt(cqW,  w_cq,     (long)CE * CE);
  cvt(ckW,  w_ck,     (long)CE * CE);
  cvt(cvW,  w_cv,     (long)CE * CE);
  cvt(l1W,  w_l1,     (long)CFF * CE);
  cvt(l2W,  w_l2,     (long)CE * CFF);

  // ---- Phase A: self attention ----
  k_add2<<<dim3((unsigned)((TOK * CE + 255) / 256)), 256, 0, stream>>>(tgt, qpos, qkin16, TOK * CE);
  cvt(tgt, tgt16, TOK * CE);

  // q/k/v projections: [2048,1024] x [1024,1024]^T
  gemm_wmma<true><<<dim3(16, 16, 1), 256, 0, stream>>>(
      qkin16, CE, 0, 0, w_inproj, CE, 0, 0, inB, 1.f, 0,
      nullptr, q16, CE, 0, 0, (int)TOK, CE, CE, 1);
  gemm_wmma<true><<<dim3(16, 16, 1), 256, 0, stream>>>(
      qkin16, CE, 0, 0, w_inproj + (long)CE * CE, CE, 0, 0, inB + CE, 1.f, 0,
      nullptr, k16, CE, 0, 0, (int)TOK, CE, CE, 1);
  gemm_wmma<true><<<dim3(16, 16, 1), 256, 0, stream>>>(
      tgt16, CE, 0, 0, w_inproj + 2L * CE * CE, CE, 0, 0, inB + 2 * CE, 1.f, 0,
      nullptr, v16, CE, 0, 0, (int)TOK, CE, CE, 1);

  // scores[b,h,l,m] = scale * q.k  (batched over z = b*H+h)
  gemm_wmma<true><<<dim3(CL / 128, CL / 64, CB * CH), 256, 0, stream>>>(
      q16, (long)CB * CE, CE, CHD,
      k16, (long)CB * CE, CE, CHD,
      nullptr, scale, 0,
      scores, nullptr, CL, (long)CH * CL * CL, (long)CL * CL,
      CL, CL, CHD, CH);

  k_softmax512<<<CB * CH * CL, 256, 0, stream>>>(scores, attn16);

  // ctx[l,b,h,:] = attn @ v   (B stored [K=m, N=d] -> strided path)
  gemm_wmma<false><<<dim3(CL / 128, 1, CB * CH), 256, 0, stream>>>(
      attn16, CL, (long)CH * CL * CL, (long)CL * CL,
      v16, (long)CB * CE, CE, CHD,
      nullptr, 1.f, 0,
      nullptr, ctx16, (long)CB * CE, CE, CHD,
      CL, CHD, CL, CH);

  // out projection + residual + LN1
  gemm_wmma<true><<<dim3(16, 16, 1), 256, 0, stream>>>(
      ctx16, CE, 0, 0, w_out, CE, 0, 0, outB, 1.f, 0,
      selfout, nullptr, CE, 0, 0, (int)TOK, CE, CE, 1);
  k_ln<<<(unsigned)TOK, 256, 0, stream>>>(tgt, selfout, nullptr, n1g, n1b, tgt1, nullptr);

  // ---- Phase B: custom cross attention ----
  k_add2<<<dim3((unsigned)((TOK * CE + 255) / 256)), 256, 0, stream>>>(tgt1, qpos, qin2, TOK * CE);
  k_add2<<<dim3((unsigned)((MTOK * CE + 255) / 256)), 256, 0, stream>>>(mem, pos, mp16, MTOK * CE);
  cvt(mem, mem16, MTOK * CE);

  gemm_wmma<true><<<dim3(16, 16, 1), 256, 0, stream>>>(
      qin2, CE, 0, 0, w_cq, CE, 0, 0, cqB, 1.f, 0,
      qc32, nullptr, CE, 0, 0, (int)TOK, CE, CE, 1);
  gemm_wmma<true><<<dim3(64, 16, 1), 256, 0, stream>>>(
      mp16, CE, 0, 0, w_ck, CE, 0, 0, ckB, 1.f, 0,
      kc32, nullptr, CE, 0, 0, (int)MTOK, CE, CE, 1);
  gemm_wmma<true><<<dim3(64, 16, 1), 256, 0, stream>>>(
      mem16, CE, 0, 0, w_cv, CE, 0, 0, cvB, 1.f, 0,
      vc32, nullptr, CE, 0, 0, (int)MTOK, CE, CE, 1);

  k_qcsum<<<(CB * CE) / 256, 256, 0, stream>>>(qc32, qcsum);
  k_s1<<<(CB * CH * CS) / 256, 256, 0, stream>>>(qcsum, kc32, s1sum, scale);
  k_cross_softmax<<<CB * CHD * CH, 256, 0, stream>>>(kc32, s1sum, prob);
  k_w2<<<(CB * CH * CS) / 256, 256, 0, stream>>>(dwW, prob, w2buf);
  k_od<<<CB * CE, 256, 0, stream>>>(w2buf, vc32, dwB, odbuf);

  k_ln<<<(unsigned)TOK, 256, 0, stream>>>(tgt1, nullptr, odbuf, n2g, n2b, tgt2, tgt2_16);

  // ---- Phase C: FFN ----
  gemm_wmma<true><<<dim3(16, 32, 1), 256, 0, stream>>>(
      tgt2_16, CE, 0, 0, w_l1, CE, 0, 0, l1B, 1.f, 1 /*relu*/,
      nullptr, h1_16, CFF, 0, 0, (int)TOK, CFF, CE, 1);
  gemm_wmma<true><<<dim3(16, 16, 1), 256, 0, stream>>>(
      h1_16, CFF, 0, 0, w_l2, CFF, 0, 0, l2B, 1.f, 0,
      ff32, nullptr, CE, 0, 0, (int)TOK, CE, CFF, 1);

  k_ln<<<(unsigned)TOK, 256, 0, stream>>>(tgt2, ff32, nullptr, n3g, n3b, (float*)d_out, nullptr);

  (void)in_sizes; (void)n_in; (void)out_size; (void)ws_size;
}
